// MultiGAT_20040317403629
// MI455X (gfx1250) — matrix operand order repaired and checked by reading, not compiled, not run on hardware
//
#include <hip/hip_runtime.h>
#include <hip/hip_bf16.h>

typedef __attribute__((ext_vector_type(16))) _Float16 v16h;
typedef __attribute__((ext_vector_type(8)))  _Float16 v8h;
typedef __attribute__((ext_vector_type(8)))  float    v8f;

#define NEG_SLOPE 0.2f
#define SM_EPS    1e-16f

// ---------- helpers: monotonic float <-> uint encoding for atomic max ----------
__device__ __forceinline__ unsigned fenc(float f) {
    int b = __float_as_int(f);
    return (b < 0) ? ~(unsigned)b : ((unsigned)b | 0x80000000u);
}
__device__ __forceinline__ float fdec(unsigned u) {
    int b = (u & 0x80000000u) ? (int)(u & 0x7FFFFFFFu) : (int)~u;
    return __int_as_float(b);
}

// ---------- pack A: fp32 [M,K] -> f16 [M,Kp] zero-padded ----------
__global__ void pack_a_f16(const float* __restrict__ A, _Float16* __restrict__ Ah,
                           int M, int K, int Kp) {
    long long i = (long long)blockIdx.x * blockDim.x + threadIdx.x;
    if (i >= (long long)M * Kp) return;
    int r = (int)(i / Kp), k = (int)(i % Kp);
    float v = (k < K) ? A[(size_t)r * K + k] : 0.0f;
    Ah[i] = (_Float16)v;
}

// ---------- pack B: fp32 [K,N] -> f16 WMMA B-fragment layout [nt][kt][lane][16] ----------
// lane L holds col = nt*16 + (L&15), halves hh -> k = kt*32 + 16*(L>>4) + hh
__global__ void pack_b_f16(const float* __restrict__ B, _Float16* __restrict__ Bp,
                           int K, int N, int Nt, int KT) {
    long long i = (long long)blockIdx.x * blockDim.x + threadIdx.x;
    long long total = (long long)Nt * KT * 32 * 16;
    if (i >= total) return;
    int hh   = (int)(i & 15);
    int lane = (int)((i >> 4) & 31);
    long long rest = i >> 9;
    int kt = (int)(rest % KT);
    int nt = (int)(rest / KT);
    int k   = kt * 32 + ((lane >> 4) << 4) + hh;
    int col = nt * 16 + (lane & 15);
    float v = (k < K && col < N) ? B[(size_t)k * N + col] : 0.0f;
    Bp[i] = (_Float16)v;
}

// ---------- WMMA GEMM on packed f16: C[M,NCOLS] = A @ B, fp32 accumulate ----------
// One wave -> one 16x16 tile. No bounds checks in K loop (pre-padded inputs).
// M must be a multiple of 16. NCHK=true iff NCOLS % 16 != 0.
template <int KT, int NCOLS, bool NCHK>
__global__ void wmma_gemm_packed(const _Float16* __restrict__ Ah,  // [M, KT*32]
                                 const _Float16* __restrict__ Bp,  // [Nt][KT][32][16]
                                 float* __restrict__ C,
                                 int M, int Nt) {
    int wave = (int)((blockIdx.x * blockDim.x + threadIdx.x) >> 5);
    int lane = (int)(threadIdx.x & 31);
    int mt = wave / Nt;
    int nt = wave - mt * Nt;
    if (mt * 16 >= M) return;                 // wave-uniform exit (grid tail)

    int g   = lane >> 4;
    int col = nt * 16 + (lane & 15);

    const v8h*  Ar = (const v8h*)(Ah + (size_t)(mt * 16 + (lane & 15)) * (KT * 32));
    const v16h* Br = (const v16h*)(Bp + ((size_t)nt * KT * 32 + lane) * 16);

    v8f acc = {};
    #pragma unroll
    for (int kt = 0; kt < KT; ++kt) {
        v8h alo = Ar[kt * 4 + 2 * g];         // K = kt*32 + 16g .. +7
        v8h ahi = Ar[kt * 4 + 2 * g + 1];     // K = kt*32 + 16g + 8 .. +15
        v16h a = __builtin_shufflevector(alo, ahi,
                    0,1,2,3,4,5,6,7,8,9,10,11,12,13,14,15);
        v16h b = Br[kt * 32];                 // contiguous 32B per lane
        acc = __builtin_amdgcn_wmma_f32_16x16x32_f16(
                  false, a, false, b, (short)0, acc, false, false);
    }
    // C fragment: VGPR r, lane L -> row = mt*16 + 8*g + r, col = col
    float* Cp = C + (size_t)(mt * 16 + g * 8) * NCOLS + col;
    if (!NCHK || col < NCOLS) {
        #pragma unroll
        for (int r = 0; r < 8; ++r)
            Cp[r * NCOLS] = acc[r];           // immediate offsets, one store clause
    }
}

// ---------- alpha_src / alpha_dst per (node, head) ----------
__global__ void alpha_kernel(const float* __restrict__ h,
                             const float* __restrict__ a_src,
                             const float* __restrict__ a_dst,
                             float* __restrict__ as_out,
                             float* __restrict__ ad_out,
                             int N, int C) {   // H == 2
    int i = blockIdx.x * blockDim.x + threadIdx.x;   // i = n*2 + head
    if (i >= N * 2) return;
    int n = i >> 1, hd = i & 1;
    const float* hp = h + (size_t)n * 2 * C + (size_t)hd * C;
    const float* av = a_src + hd * C;
    const float* bv = a_dst + hd * C;
    float s1 = 0.0f, s2 = 0.0f;
    #pragma unroll 4
    for (int c = 0; c < C; ++c) {
        float v = hp[c];
        s1 += v * av[c];
        s2 += v * bv[c];
    }
    as_out[i] = s1;
    ad_out[i] = s2;
}

// ---------- edge pass 1: leaky-relu logits + segment max (encoded atomics) ----------
__global__ void edge_max_kernel(const long long* __restrict__ srcI,
                                const long long* __restrict__ dstI,
                                int E, int N,
                                const float* __restrict__ as,
                                const float* __restrict__ ad,
                                unsigned* __restrict__ menc) {
    int e = blockIdx.x * blockDim.x + threadIdx.x;
    int ET = E + N;
    if (e >= ET) return;
    int s = (e < E) ? (int)srcI[e] : (e - E);
    int d = (e < E) ? (int)dstI[e] : (e - E);
    #pragma unroll
    for (int hd = 0; hd < 2; ++hd) {
        float v = as[s * 2 + hd] + ad[d * 2 + hd];
        v = (v > 0.0f) ? v : NEG_SLOPE * v;
        atomicMax(&menc[d * 2 + hd], fenc(v));
    }
}

// ---------- edge pass 2: p = exp(e - m), segment sum ----------
__global__ void edge_exp_kernel(const long long* __restrict__ srcI,
                                const long long* __restrict__ dstI,
                                int E, int N,
                                const float* __restrict__ as,
                                const float* __restrict__ ad,
                                const unsigned* __restrict__ menc,
                                float* __restrict__ p,
                                float* __restrict__ ssum) {
    int e = blockIdx.x * blockDim.x + threadIdx.x;
    int ET = E + N;
    if (e >= ET) return;
    int s = (e < E) ? (int)srcI[e] : (e - E);
    int d = (e < E) ? (int)dstI[e] : (e - E);
    #pragma unroll
    for (int hd = 0; hd < 2; ++hd) {
        float v = as[s * 2 + hd] + ad[d * 2 + hd];
        v = (v > 0.0f) ? v : NEG_SLOPE * v;
        float m  = fdec(menc[d * 2 + hd]);
        float pe = __expf(v - m);
        p[(size_t)e * 2 + hd] = pe;
        atomicAdd(&ssum[d * 2 + hd], pe);
    }
}

// ---------- edge pass 3: agg[dst] += h[src] * alpha  (one wave per edge) ----------
__global__ void edge_agg_kernel(const long long* __restrict__ srcI,
                                const long long* __restrict__ dstI,
                                int E, int N,
                                const float* __restrict__ h,
                                const float* __restrict__ p,
                                const float* __restrict__ ssum,
                                float* __restrict__ agg,
                                int C, int F) {            // F = 2*C
    int wave = (int)((blockIdx.x * blockDim.x + threadIdx.x) >> 5);
    int lane = (int)(threadIdx.x & 31);
    int ET = E + N;
    if (wave >= ET) return;
    int e = wave;
    int s = (e < E) ? (int)srcI[e] : (e - E);
    int d = (e < E) ? (int)dstI[e] : (e - E);
    float a0 = p[(size_t)e * 2 + 0] / (ssum[d * 2 + 0] + SM_EPS);
    float a1 = p[(size_t)e * 2 + 1] / (ssum[d * 2 + 1] + SM_EPS);
    const float* hs = h   + (size_t)s * F;
    float*       od = agg + (size_t)d * F;
    #pragma unroll 4
    for (int f = lane; f < F; f += 32) {
        float al = (f < C) ? a0 : a1;
        atomicAdd(&od[f], hs[f] * al);
    }
}

// ---------- epilogue: bias (+ optional ELU), in place ----------
__global__ void epilogue_kernel(float* __restrict__ x,
                                const float* __restrict__ b,
                                int N, int F, int do_elu) {
    int i = blockIdx.x * blockDim.x + threadIdx.x;
    if (i >= N * F) return;
    float v = x[i] + b[i % F];
    if (do_elu) v = (v > 0.0f) ? v : (__expf(v) - 1.0f);
    x[i] = v;
}

static inline int cdiv(long long a, long long b) { return (int)((a + b - 1) / b); }

extern "C" void kernel_launch(void* const* d_in, const int* in_sizes, int n_in,
                              void* d_out, int out_size, void* d_ws, size_t ws_size,
                              hipStream_t stream) {
    const float*     x      = (const float*)d_in[0];
    const long long* edges  = (const long long*)d_in[1];
    const float*     W0     = (const float*)d_in[2];
    const float*     a_src0 = (const float*)d_in[3];
    const float*     a_dst0 = (const float*)d_in[4];
    const float*     b0     = (const float*)d_in[5];
    const float*     W1     = (const float*)d_in[6];
    const float*     a_src1 = (const float*)d_in[7];
    const float*     a_dst1 = (const float*)d_in[8];
    const float*     b1     = (const float*)d_in[9];

    const int N = in_sizes[0] / 17;        // 50000
    const int E = in_sizes[1] / 2;         // 800000
    const int ET = E + N;                  // with self loops
    const long long* srcI = edges;         // edges[0,:]
    const long long* dstI = edges + E;     // edges[1,:]

    const int F0 = 256, C0 = 128;          // layer 0: 2 heads x 128
    const int F1 = 200, C1 = 100;          // layer 1: 2 heads x 100
    const int Nt0 = 16, Nt1 = 13;          // N tiles (16-wide)

    // ---- workspace carve-up ----
    float*    h    = (float*)d_ws;                         // N*256 f32
    float*    agg  = h   + (size_t)N * 256;                // N*256 f32 (agg0 -> x1)
    float*    asv  = agg + (size_t)N * 256;                // N*2
    float*    adv  = asv + (size_t)N * 2;                  // N*2
    unsigned* menc = (unsigned*)(adv + (size_t)N * 2);     // N*2
    float*    ssum = (float*)(menc + (size_t)N * 2);       // N*2
    float*    p    = ssum + (size_t)N * 2;                 // ET*2
    _Float16* Ah   = (_Float16*)(p + (size_t)ET * 2);      // N*256 f16 (packed A)
    _Float16* Bp   = Ah + (size_t)N * 256;                 // <=13*8*512 f16 (packed B)
    float*    out  = (float*)d_out;                        // N*200

    const int TB = 256;

    // ================= Layer 0 =================
    hipMemsetAsync(agg,  0, (size_t)N * 256 * sizeof(float), stream);
    hipMemsetAsync(menc, 0, (size_t)N * 2 * sizeof(unsigned), stream);
    hipMemsetAsync(ssum, 0, (size_t)N * 2 * sizeof(float), stream);

    {   // h = x @ W0   (M=N, K=17 -> Kp=32, Ncols=256)
        pack_a_f16<<<cdiv((long long)N * 32, TB), TB, 0, stream>>>(x, Ah, N, 17, 32);
        pack_b_f16<<<cdiv((long long)Nt0 * 1 * 512, TB), TB, 0, stream>>>(W0, Bp, 17, F0, Nt0, 1);
        long long waves = (long long)cdiv(N, 16) * Nt0;
        wmma_gemm_packed<1, 256, false><<<cdiv(waves * 32, TB), TB, 0, stream>>>(Ah, Bp, h, N, Nt0);
    }
    alpha_kernel<<<cdiv((long long)N * 2, TB), TB, 0, stream>>>(h, a_src0, a_dst0, asv, adv, N, C0);
    edge_max_kernel<<<cdiv(ET, TB), TB, 0, stream>>>(srcI, dstI, E, N, asv, adv, menc);
    edge_exp_kernel<<<cdiv(ET, TB), TB, 0, stream>>>(srcI, dstI, E, N, asv, adv, menc, p, ssum);
    edge_agg_kernel<<<cdiv((long long)ET * 32, TB), TB, 0, stream>>>(srcI, dstI, E, N, h, p, ssum, agg, C0, F0);
    epilogue_kernel<<<cdiv((long long)N * F0, TB), TB, 0, stream>>>(agg, b0, N, F0, 1);
    // agg now holds x1 = elu(agg0 + b0)

    // ================= Layer 1 =================
    hipMemsetAsync(menc, 0, (size_t)N * 2 * sizeof(unsigned), stream);
    hipMemsetAsync(ssum, 0, (size_t)N * 2 * sizeof(float), stream);
    hipMemsetAsync(out,  0, (size_t)N * F1 * sizeof(float), stream);

    {   // h = x1 @ W1  (M=N, K=256 -> KT=8, Ncols=200 -> Nt=13)
        pack_a_f16<<<cdiv((long long)N * 256, TB), TB, 0, stream>>>(agg, Ah, N, 256, 256);
        pack_b_f16<<<cdiv((long long)Nt1 * 8 * 512, TB), TB, 0, stream>>>(W1, Bp, 256, F1, Nt1, 8);
        long long waves = (long long)cdiv(N, 16) * Nt1;
        wmma_gemm_packed<8, 200, true><<<cdiv(waves * 32, TB), TB, 0, stream>>>(Ah, Bp, h, N, Nt1);
    }
    alpha_kernel<<<cdiv((long long)N * 2, TB), TB, 0, stream>>>(h, a_src1, a_dst1, asv, adv, N, C1);
    edge_max_kernel<<<cdiv(ET, TB), TB, 0, stream>>>(srcI, dstI, E, N, asv, adv, menc);
    edge_exp_kernel<<<cdiv(ET, TB), TB, 0, stream>>>(srcI, dstI, E, N, asv, adv, menc, p, ssum);
    edge_agg_kernel<<<cdiv((long long)ET * 32, TB), TB, 0, stream>>>(srcI, dstI, E, N, h, p, ssum, out, C1, F1);
    epilogue_kernel<<<cdiv((long long)N * F1, TB), TB, 0, stream>>>(out, b1, N, F1, 0);
}
